// MSDeformAttnTransformerEncoderOnly_19258633355697
// MI455X (gfx1250) — matrix operand order repaired and checked by reading, not compiled, not run on hardware
//
#include <hip/hip_runtime.h>
#include <hip/hip_bf16.h>
#include <stdint.h>

// ---------------- problem constants ----------------
#define NLAYERS 6
#define DMODEL  256
#define NHEAD   8
#define DHEAD   32
#define NLVL    4
#define NPNT    4
#define DFF     1024
#define BATCH   4
#define LQTOT   5440            // 64*64 + 32*32 + 16*16 + 8*8
#define BLQ     (BATCH * LQTOT) // 21760  (== 170 * 128)

typedef __attribute__((ext_vector_type(16))) __bf16 v16bf;
typedef __attribute__((ext_vector_type(8)))  __bf16 v8bf;
typedef __attribute__((ext_vector_type(8)))  float  v8f;

// =====================================================================
// Weight convert: f32 [L][K][N] -> bf16 transposed [L][N][K]
// =====================================================================
__global__ void wconv_kernel(const float* __restrict__ src, __bf16* __restrict__ dst,
                             int K, int N)
{
    size_t i = (size_t)blockIdx.x * blockDim.x + threadIdx.x;
    size_t per = (size_t)K * N;
    size_t tot = per * NLAYERS;
    if (i >= tot) return;
    size_t lyr = i / per;
    size_t r   = i - lyr * per;
    size_t n   = r / K;
    size_t k   = r - n * K;
    dst[i] = (__bf16)src[lyr * per + k * (size_t)N + n];
}

// =====================================================================
// Flatten multi-scale maps: src[B,D,h,w] -> out[B,LQ,D]; pos + level_embed
// =====================================================================
__global__ void pack_kernel(const float* __restrict__ s0, const float* __restrict__ p0,
                            const float* __restrict__ s1, const float* __restrict__ p1,
                            const float* __restrict__ s2, const float* __restrict__ p2,
                            const float* __restrict__ s3, const float* __restrict__ p3,
                            const float* __restrict__ lvl_emb,
                            float* __restrict__ out, float* __restrict__ pos)
{
    int bq = blockIdx.x;            // 0 .. BLQ-1
    int d  = threadIdx.x;           // 0 .. 255
    int b  = bq / LQTOT;
    int lq = bq - b * LQTOT;
    int l, s, hw;
    const float *sp, *pp;
    if      (lq < 4096) { l = 0; s = lq;        hw = 4096; sp = s0; pp = p0; }
    else if (lq < 5120) { l = 1; s = lq - 4096; hw = 1024; sp = s1; pp = p1; }
    else if (lq < 5376) { l = 2; s = lq - 5120; hw = 256;  sp = s2; pp = p2; }
    else                { l = 3; s = lq - 5376; hw = 64;   sp = s3; pp = p3; }
    size_t si = ((size_t)b * DMODEL + d) * hw + s;
    size_t di = (size_t)bq * DMODEL + d;
    out[di] = sp[si];
    pos[di] = pp[si] + lvl_emb[l * DMODEL + d];
}

// =====================================================================
// q = out + pos (bf16), out (bf16)
// =====================================================================
__global__ void qprep_kernel(const float* __restrict__ out, const float* __restrict__ pos,
                             __bf16* __restrict__ qb, __bf16* __restrict__ outb, size_t n)
{
    size_t i = (size_t)blockIdx.x * blockDim.x + threadIdx.x;
    if (i >= n) return;
    float o = out[i];
    qb[i]   = (__bf16)(o + pos[i]);
    outb[i] = (__bf16)o;
}

// =====================================================================
// WMMA bf16 GEMM:  C[M,N] = A[M,K] * Bt[N,K]^T + bias
//   block = 256 threads = 8 waves: 4 along M x 2 along N -> 128M x 128N tile
//   wave  = 32M x 64N (2 A frags x 4 B frags -> 8 accumulators),
//   K step 32 via v_wmma_f32_16x16x32_bf16
// A fragment layout (16-bit A 16x32): lane m=lane&15; lanes 0-15 hold
//   K=0..15, lanes 16-31 hold K=16..31 (contiguous in A row).
// B fragment layout (16-bit B 32x16): lane n=lane&15; lanes 0-15 hold K=0..15,
//   lanes 16-31 hold K=16..31 (contiguous in Bt row).
// =====================================================================
__launch_bounds__(256, 2)
__global__ void gemm_wmma_bf16(const __bf16* __restrict__ A,
                               const __bf16* __restrict__ Bt,
                               const float*  __restrict__ bias,
                               float*        __restrict__ Cf,
                               __bf16*       __restrict__ Cb,
                               int M, int N, int K, int relu)
{
    const int lane  = threadIdx.x & 31;
    const int wave  = threadIdx.x >> 5;
    const int wm    = wave & 3;
    const int wn    = wave >> 2;
    const int mbase = blockIdx.y * 128 + wm * 32;
    const int nbase = blockIdx.x * 128 + wn * 64;
    const int mrow  = lane & 15;
    const int half  = lane >> 4;

    const __bf16* arow0 = A  + (size_t)(mbase + mrow) * K + half * 16;
    const __bf16* arow1 = arow0 + (size_t)16 * K;
    const __bf16* brow0 = Bt + (size_t)(nbase + mrow) * K + half * 16;
    const __bf16* brow1 = brow0 + (size_t)16 * K;
    const __bf16* brow2 = brow0 + (size_t)32 * K;
    const __bf16* brow3 = brow0 + (size_t)48 * K;

    v8f acc00 = {}, acc01 = {}, acc02 = {}, acc03 = {};
    v8f acc10 = {}, acc11 = {}, acc12 = {}, acc13 = {};

    for (int k0 = 0; k0 < K; k0 += 32) {
        v8bf a0lo = *(const v8bf*)(arow0 + k0);
        v8bf a0hi = *(const v8bf*)(arow0 + k0 + 8);
        v8bf a1lo = *(const v8bf*)(arow1 + k0);
        v8bf a1hi = *(const v8bf*)(arow1 + k0 + 8);
        v16bf a0  = __builtin_shufflevector(a0lo, a0hi,
                      0,1,2,3,4,5,6,7,8,9,10,11,12,13,14,15);
        v16bf a1  = __builtin_shufflevector(a1lo, a1hi,
                      0,1,2,3,4,5,6,7,8,9,10,11,12,13,14,15);
        v16bf b0 = *(const v16bf*)(brow0 + k0);
        v16bf b1 = *(const v16bf*)(brow1 + k0);
        v16bf b2 = *(const v16bf*)(brow2 + k0);
        v16bf b3 = *(const v16bf*)(brow3 + k0);
        acc00 = __builtin_amdgcn_wmma_f32_16x16x32_bf16(false, a0, false, b0, (short)0, acc00, false, false);
        acc01 = __builtin_amdgcn_wmma_f32_16x16x32_bf16(false, a0, false, b1, (short)0, acc01, false, false);
        acc02 = __builtin_amdgcn_wmma_f32_16x16x32_bf16(false, a0, false, b2, (short)0, acc02, false, false);
        acc03 = __builtin_amdgcn_wmma_f32_16x16x32_bf16(false, a0, false, b3, (short)0, acc03, false, false);
        acc10 = __builtin_amdgcn_wmma_f32_16x16x32_bf16(false, a1, false, b0, (short)0, acc10, false, false);
        acc11 = __builtin_amdgcn_wmma_f32_16x16x32_bf16(false, a1, false, b1, (short)0, acc11, false, false);
        acc12 = __builtin_amdgcn_wmma_f32_16x16x32_bf16(false, a1, false, b2, (short)0, acc12, false, false);
        acc13 = __builtin_amdgcn_wmma_f32_16x16x32_bf16(false, a1, false, b3, (short)0, acc13, false, false);
    }

    // Epilogue: C layout per 16x16 f32 tile -> row m = r + 8*half, col n = lane&15
#define EPILOG(ACC, MI, T)                                                     \
    {                                                                          \
        int n = nbase + (T)*16 + mrow;                                         \
        float bv = bias ? bias[n] : 0.f;                                       \
        _Pragma("unroll")                                                      \
        for (int r = 0; r < 8; ++r) {                                          \
            float val = ACC[r] + bv;                                           \
            if (relu) val = fmaxf(val, 0.f);                                   \
            size_t idx = (size_t)(mbase + (MI)*16 + r + half * 8) * N + n;     \
            if (Cf) Cf[idx] = val;                                             \
            if (Cb) Cb[idx] = (__bf16)val;                                     \
        }                                                                      \
    }
    EPILOG(acc00, 0, 0)
    EPILOG(acc01, 0, 1)
    EPILOG(acc02, 0, 2)
    EPILOG(acc03, 0, 3)
    EPILOG(acc10, 1, 0)
    EPILOG(acc11, 1, 1)
    EPILOG(acc12, 1, 2)
    EPILOG(acc13, 1, 3)
#undef EPILOG
}

// =====================================================================
// Softmax over 16 (NL*NP) per (query, head)
// =====================================================================
__global__ void softmax16_kernel(float* __restrict__ aw, int total)
{
    int i = blockIdx.x * blockDim.x + threadIdx.x;
    if (i >= total) return;
    float* p = aw + (size_t)i * 16;
    float m = p[0];
#pragma unroll
    for (int j = 1; j < 16; ++j) m = fmaxf(m, p[j]);
    float e[16];
    float s = 0.f;
#pragma unroll
    for (int j = 0; j < 16; ++j) { e[j] = __expf(p[j] - m); s += e[j]; }
    float inv = 1.f / s;
#pragma unroll
    for (int j = 0; j < 16; ++j) p[j] = e[j] * inv;
}

// =====================================================================
// Deformable attention sampling. One wave per (b, lq, head); lane = channel.
// off: [BLQ, NH*NL*NP*2] (h-major), aw: [BLQ, NH*NL*NP], v: [BLQ, D] (h,c)
// Output acc (bf16) in [BLQ, D] layout, ready as GEMM A operand for Wo.
// =====================================================================
__launch_bounds__(256)
__global__ void msdeform_kernel(const float* __restrict__ vbuf,
                                const float* __restrict__ off,
                                const float* __restrict__ aw,
                                __bf16* __restrict__ accb)
{
    const int lane = threadIdx.x & 31;
    int wid = (blockIdx.x * blockDim.x + threadIdx.x) >> 5;   // [0, BLQ*NH)
    int h   = wid & (NHEAD - 1);
    int bq  = wid >> 3;
    int b   = bq / LQTOT;
    int lq  = bq - b * LQTOT;

    // reference point of this query (levels are square: h == w)
    int qw, qs;
    if      (lq < 4096) { qs = lq;        qw = 64; }
    else if (lq < 5120) { qs = lq - 4096; qw = 32; }
    else if (lq < 5376) { qs = lq - 5120; qw = 16; }
    else                { qs = lq - 5376; qw = 8;  }
    float rx = ((qs % qw) + 0.5f) / (float)qw;
    float ry = ((qs / qw) + 0.5f) / (float)qw;

    const float* offp = off + (size_t)bq * (NHEAD * NLVL * NPNT * 2) + h * (NLVL * NPNT * 2);
    const float* awp  = aw  + (size_t)bq * (NHEAD * NLVL * NPNT)     + h * (NLVL * NPNT);

    const int LDIM[4]   = {64, 32, 16, 8};
    const int LSTART[4] = {0, 4096, 5120, 5376};

    float acc = 0.f;
#pragma unroll
    for (int l = 0; l < NLVL; ++l) {
        const int   W    = LDIM[l];
        const float Wf   = (float)W;
        const float invW = 1.f / Wf;
        const float* vl  = vbuf + ((size_t)b * LQTOT + LSTART[l]) * DMODEL + h * DHEAD + lane;
#pragma unroll
        for (int p = 0; p < NPNT; ++p) {
            float ox = offp[l * 8 + p * 2 + 0];
            float oy = offp[l * 8 + p * 2 + 1];
            float a  = awp[l * 4 + p];
            float x  = (rx + ox * invW) * Wf - 0.5f;
            float y  = (ry + oy * invW) * Wf - 0.5f;
            float xf = floorf(x), yf = floorf(y);
            float wx = x - xf,    wy = y - yf;
            int   x0 = (int)xf,   y0 = (int)yf;
            float sum = 0.f;
#pragma unroll
            for (int dy = 0; dy < 2; ++dy) {
#pragma unroll
                for (int dx = 0; dx < 2; ++dx) {
                    int xi = x0 + dx, yi = y0 + dy;
                    if (xi >= 0 && xi < W && yi >= 0 && yi < W) {
                        float wgt = (dx ? wx : 1.f - wx) * (dy ? wy : 1.f - wy);
                        sum += wgt * vl[(size_t)(yi * W + xi) * DMODEL];
                    }
                }
            }
            acc += a * sum;
        }
    }
    accb[(size_t)bq * DMODEL + h * DHEAD + lane] = (__bf16)acc;
}

// =====================================================================
// Fused residual + layernorm. One wave per row (256 channels, 8 per lane).
// Writes f32 (next residual stream) and bf16 (next GEMM A operand).
// =====================================================================
__launch_bounds__(256)
__global__ void add_ln_kernel(const float* __restrict__ x, const float* __restrict__ r,
                              const float* __restrict__ g, const float* __restrict__ bta,
                              float* __restrict__ outf, __bf16* __restrict__ outb)
{
    const int lane = threadIdx.x & 31;
    int row = (blockIdx.x * blockDim.x + threadIdx.x) >> 5;   // [0, BLQ)
    const float* xp = x + (size_t)row * DMODEL;
    const float* rp = r + (size_t)row * DMODEL;

    float v[8];
    float s = 0.f;
#pragma unroll
    for (int j = 0; j < 8; ++j) {
        v[j] = xp[lane + j * 32] + rp[lane + j * 32];
        s += v[j];
    }
#pragma unroll
    for (int o = 16; o; o >>= 1) s += __shfl_xor(s, o, 32);
    float mean = s * (1.f / 256.f);

    float vs = 0.f;
#pragma unroll
    for (int j = 0; j < 8; ++j) { float d = v[j] - mean; vs += d * d; }
#pragma unroll
    for (int o = 16; o; o >>= 1) vs += __shfl_xor(vs, o, 32);
    float rstd = rsqrtf(vs * (1.f / 256.f) + 1e-5f);

#pragma unroll
    for (int j = 0; j < 8; ++j) {
        int c = lane + j * 32;
        float y = (v[j] - mean) * rstd * g[c] + bta[c];
        size_t idx = (size_t)row * DMODEL + c;
        outf[idx] = y;
        outb[idx] = (__bf16)y;
    }
}

// =====================================================================
// host-side driver
// =====================================================================
extern "C" void kernel_launch(void* const* d_in, const int* in_sizes, int n_in,
                              void* d_out, int out_size, void* d_ws, size_t ws_size,
                              hipStream_t stream)
{
    // setup_inputs() dict order
    const float* src0 = (const float*)d_in[0];
    const float* pos0 = (const float*)d_in[1];
    const float* src1 = (const float*)d_in[2];
    const float* pos1 = (const float*)d_in[3];
    const float* src2 = (const float*)d_in[4];
    const float* pos2 = (const float*)d_in[5];
    const float* src3 = (const float*)d_in[6];
    const float* pos3 = (const float*)d_in[7];
    const float* lvl  = (const float*)d_in[8];
    const float* W_off = (const float*)d_in[9];
    const float* b_off = (const float*)d_in[10];
    const float* W_aw  = (const float*)d_in[11];
    const float* b_aw  = (const float*)d_in[12];
    const float* Wv    = (const float*)d_in[13];
    const float* bv    = (const float*)d_in[14];
    const float* Wo    = (const float*)d_in[15];
    const float* bo    = (const float*)d_in[16];
    const float* ln1g  = (const float*)d_in[17];
    const float* ln1b  = (const float*)d_in[18];
    const float* W1    = (const float*)d_in[19];
    const float* b1    = (const float*)d_in[20];
    const float* W2    = (const float*)d_in[21];
    const float* b2    = (const float*)d_in[22];
    const float* ln2g  = (const float*)d_in[23];
    const float* ln2b  = (const float*)d_in[24];

    // ---- workspace carve-out ----
    char* ws = (char*)d_ws;
    auto take = [&](size_t bytes) -> void* {
        void* p = (void*)ws;
        ws += (bytes + 255) & ~(size_t)255;
        return p;
    };
    float*  outbuf   = (float*) take((size_t)BLQ * DMODEL * 4);
    float*  posbuf   = (float*) take((size_t)BLQ * DMODEL * 4);
    float*  scratchN = (float*) take((size_t)BLQ * DMODEL * 4);  // off -> src2 -> ffn
    float*  awbuf    = (float*) take((size_t)BLQ * 128 * 4);
    float*  vbuf     = (float*) take((size_t)BLQ * DMODEL * 4);
    __bf16* qb       = (__bf16*)take((size_t)BLQ * DMODEL * 2);
    __bf16* outb     = (__bf16*)take((size_t)BLQ * DMODEL * 2);
    __bf16* accb     = (__bf16*)take((size_t)BLQ * DMODEL * 2);
    __bf16* hidb     = (__bf16*)take((size_t)BLQ * DFF * 2);
    __bf16* WoffT    = (__bf16*)take((size_t)NLAYERS * 256 * 256 * 2);
    __bf16* WawT     = (__bf16*)take((size_t)NLAYERS * 128 * 256 * 2);
    __bf16* WvT      = (__bf16*)take((size_t)NLAYERS * 256 * 256 * 2);
    __bf16* WoT      = (__bf16*)take((size_t)NLAYERS * 256 * 256 * 2);
    __bf16* W1T      = (__bf16*)take((size_t)NLAYERS * 1024 * 256 * 2);
    __bf16* W2T      = (__bf16*)take((size_t)NLAYERS * 256 * 1024 * 2);
    (void)ws_size; (void)in_sizes; (void)n_in; (void)out_size;

    // ---- weight transpose + bf16 convert (each call; deterministic) ----
    auto lconv = [&](const float* s, __bf16* d, int K, int N) {
        size_t tot = (size_t)NLAYERS * K * N;
        wconv_kernel<<<(unsigned)((tot + 255) / 256), 256, 0, stream>>>(s, d, K, N);
    };
    lconv(W_off, WoffT, 256, 256);
    lconv(W_aw,  WawT,  256, 128);
    lconv(Wv,    WvT,   256, 256);
    lconv(Wo,    WoT,   256, 256);
    lconv(W1,    W1T,   256, 1024);
    lconv(W2,    W2T,  1024, 256);

    // ---- flatten inputs ----
    pack_kernel<<<BLQ, DMODEL, 0, stream>>>(src0, pos0, src1, pos1, src2, pos2,
                                            src3, pos3, lvl, outbuf, posbuf);

    const size_t nelem = (size_t)BLQ * DMODEL;
    const dim3 gN256(2, 170), gN128(1, 170), gN1024(8, 170);

    for (int i = 0; i < NLAYERS; ++i) {
        // q = out + pos (bf16); out (bf16)
        qprep_kernel<<<(unsigned)((nelem + 255) / 256), 256, 0, stream>>>(
            outbuf, posbuf, qb, outb, nelem);

        // off = q @ W_off + b_off   [BLQ, 256]
        gemm_wmma_bf16<<<gN256, 256, 0, stream>>>(
            qb, WoffT + (size_t)i * 256 * 256, b_off + (size_t)i * 256,
            scratchN, nullptr, BLQ, 256, 256, 0);
        // aw_logits = q @ W_aw + b_aw   [BLQ, 128]
        gemm_wmma_bf16<<<gN128, 256, 0, stream>>>(
            qb, WawT + (size_t)i * 128 * 256, b_aw + (size_t)i * 128,
            awbuf, nullptr, BLQ, 128, 256, 0);
        // v = out @ Wv + bv   [BLQ, 256]
        gemm_wmma_bf16<<<gN256, 256, 0, stream>>>(
            outb, WvT + (size_t)i * 256 * 256, bv + (size_t)i * 256,
            vbuf, nullptr, BLQ, 256, 256, 0);

        // softmax over NL*NP per head
        softmax16_kernel<<<(BLQ * NHEAD + 255) / 256, 256, 0, stream>>>(awbuf, BLQ * NHEAD);

        // deformable sampling -> acc (bf16)
        msdeform_kernel<<<BLQ, 256, 0, stream>>>(vbuf, scratchN, awbuf, accb);

        // src2 = acc @ Wo + bo  -> scratchN (off no longer needed)
        gemm_wmma_bf16<<<gN256, 256, 0, stream>>>(
            accb, WoT + (size_t)i * 256 * 256, bo + (size_t)i * 256,
            scratchN, nullptr, BLQ, 256, 256, 0);

        // out = LN1(out + src2)
        add_ln_kernel<<<BLQ / 8, 256, 0, stream>>>(
            outbuf, scratchN, ln1g + (size_t)i * 256, ln1b + (size_t)i * 256,
            outbuf, outb);

        // hid = relu(out @ W1 + b1)  (bf16 only)
        gemm_wmma_bf16<<<gN1024, 256, 0, stream>>>(
            outb, W1T + (size_t)i * 1024 * 256, b1 + (size_t)i * 1024,
            nullptr, hidb, BLQ, 1024, 256, 1);
        // ffn = hid @ W2 + b2 -> scratchN
        gemm_wmma_bf16<<<gN256, 256, 0, stream>>>(
            hidb, W2T + (size_t)i * 256 * 1024, b2 + (size_t)i * 256,
            scratchN, nullptr, BLQ, 256, 1024, 0);

        // out = LN2(out + ffn); last layer writes d_out
        float* dst = (i == NLAYERS - 1) ? (float*)d_out : outbuf;
        add_ln_kernel<<<BLQ / 8, 256, 0, stream>>>(
            outbuf, scratchN, ln2g + (size_t)i * 256, ln2b + (size_t)i * 256,
            dst, outb);
    }
}
